// Attention_21457656610958
// MI455X (gfx1250) — matrix operand order repaired and checked by reading, not compiled, not run on hardware
//
#include <hip/hip_runtime.h>
#include <hip/hip_bf16.h>

// ---------------------------------------------------------------------------
// Attention: out = softmaxish(|xWq| |xWk|^T / sqrt(D)) @ |xWv|,  N=8192, D=512
// Global-max shift cancels in A/norm => flash-style running row max is exact
// (up to EPS=1e-8 in the denominator).
// Data layout choices make every WMMA fragment a contiguous per-lane load:
//   - W stored transposed [n][k]  -> QKV B-frags contiguous (global)
//   - Q,K stored row-major [n][d] -> S A/B-frags contiguous (global)
//   - V stored transposed [d][n]  -> PV B-frags contiguous (global)
// ---------------------------------------------------------------------------

#define NN 8192
#define DD 512

// workspace layout in bf16 (ushort) elements
#define OFF_X   0u          // x bf16            8192*512
#define OFF_WQ  4194304u    // Wq^T bf16 [n][k]  512*512
#define OFF_WK  4456448u    // Wk^T
#define OFF_WV  4718592u    // Wv^T
#define OFF_Q   4980736u    // Q bf16 row-major [N,D]
#define OFF_K   9175040u    // K bf16 row-major [N,D]
#define OFF_V   13369344u   // V bf16 TRANSPOSED [D,N]

typedef __attribute__((ext_vector_type(16))) __bf16          v16bf;
typedef __attribute__((ext_vector_type(16))) unsigned short  v16u;
typedef __attribute__((ext_vector_type(8)))  unsigned short  v8u;
typedef __attribute__((ext_vector_type(8)))  float           v8f;

union BF16Frag {
    v16u u;       // element view
    v16bf b;      // WMMA operand view
    v8u  h[2];    // two 16-byte chunks -> b128 loads
};

__device__ __forceinline__ unsigned short f2bf(float f) {
    unsigned int u = __float_as_uint(f);
    u += 0x7FFFu + ((u >> 16) & 1u);          // round-to-nearest-even
    return (unsigned short)(u >> 16);
}

__device__ __forceinline__ v8f wmma_bf16(v16bf a, v16bf b, v8f c) {
    return __builtin_amdgcn_wmma_f32_16x16x32_bf16(
        /*neg_a=*/false, a, /*neg_b=*/false, b,
        /*c_mod=*/(short)0, c, /*reuse_a=*/false, /*reuse_b=*/false);
}

// ---------------------------------------------------------------------------
// Kernel 0: fp32 -> bf16; x copied straight, W matrices stored TRANSPOSED
// ---------------------------------------------------------------------------
__global__ void cvt_kernel(const float* __restrict__ x,
                           const float* __restrict__ wq,
                           const float* __restrict__ wk,
                           const float* __restrict__ wv,
                           unsigned short* __restrict__ ws) {
    const unsigned int stride = gridDim.x * blockDim.x;
    const unsigned int gid = blockIdx.x * blockDim.x + threadIdx.x;

    for (unsigned int i = gid; i < 4194304u; i += stride)          // x
        ws[OFF_X + i] = f2bf(x[i]);

    for (unsigned int j = gid; j < 786432u; j += stride) {         // W^T
        unsigned int m = j >> 18;            // matrix id 0..2
        unsigned int r = j & 262143u;        // source linear = k*512 + n
        unsigned int k = r >> 9, n = r & 511u;
        const float* src = (m == 0) ? wq : (m == 1) ? wk : wv;
        ws[OFF_WQ + m * 262144u + n * DD + k] = f2bf(src[r]);
    }
}

// ---------------------------------------------------------------------------
// Kernel 1: Q/K/V = |x @ W|.   grid = (N/16, D/128, 3), block = 256 (8 waves)
// LDS-free / barrier-free: all fragments are contiguous global vector loads
// (x rows and W^T rows are L2/WGP$ resident).
// ---------------------------------------------------------------------------
__global__ __launch_bounds__(256) void qkv_kernel(unsigned short* __restrict__ ws) {
    const unsigned short* xb = ws + OFF_X;
    const unsigned short* Wt = ws + OFF_WQ + blockIdx.z * 262144u;
    unsigned short* outQ  = ws + OFF_Q;
    unsigned short* outK  = ws + OFF_K;
    unsigned short* outVt = ws + OFF_V;

    const int m0   = blockIdx.x * 16;
    const int n0   = blockIdx.y * 128;
    const int tid  = threadIdx.x;
    const int wave = tid >> 5;
    const int lane = tid & 31;
    const int h    = lane >> 4;   // lane half
    const int nloc = lane & 15;

    const unsigned short* xrow = xb + (unsigned)(m0 + nloc) * DD;           // A row
    const unsigned short* wrow = Wt + (unsigned)(n0 + wave * 16 + nloc) * DD; // B col

    v8f acc = {};
    for (int k0 = 0; k0 < DD; k0 += 32) {
        BF16Frag a, b;
        a.h[0] = *(const v8u*)(xrow + k0 + 16 * h);         // K = 16h..16h+7
        a.h[1] = *(const v8u*)(xrow + k0 + 16 * h + 8);     // K = 16h+8..
        b.u    = *(const v16u*)(wrow + k0 + 16 * h);        // K = 16h..16h+15
        acc = wmma_bf16(a.b, b.b, acc);
    }

#pragma unroll
    for (int j = 0; j < 8; ++j) {               // C: row = j + 8h, col = lane&15
        int row = m0 + j + 8 * h;
        int col = n0 + wave * 16 + nloc;
        unsigned short bv = f2bf(fabsf(acc[j]));
        if      (blockIdx.z == 0) outQ[row * DD + col] = bv;
        else if (blockIdx.z == 1) outK[row * DD + col] = bv;
        else                      outVt[col * NN + row] = bv;  // transposed
    }
}

// ---------------------------------------------------------------------------
// Kernel 2: flash attention over a 16-row query block. block = 256 (8 waves).
// Q/K/V fragments straight from global (L2/WGP$ hot); only S/P tiles in LDS.
// ---------------------------------------------------------------------------
__global__ __launch_bounds__(256) void flash_kernel(const unsigned short* __restrict__ ws,
                                                    float* __restrict__ out) {
    const unsigned short* Qb = ws + OFF_Q;
    const unsigned short* Kb = ws + OFF_K;
    const unsigned short* Vt = ws + OFF_V;

    const int m0   = blockIdx.x * 16;
    const int tid  = threadIdx.x;
    const int wave = tid >> 5;
    const int lane = tid & 31;
    const int h    = lane >> 4;
    const int nloc = lane & 15;

    __shared__ float          sS[16 * 128];     // 8 KB
    __shared__ unsigned short sP[16 * 128];     // 4 KB
    __shared__ float sredm[16][16];             // row-max partials
    __shared__ float sreds[16][16];             // row-sum partials
    __shared__ float sm[16], sl[16], sscale[16];

    const float rs = 0.04419417382415922f;      // 1/sqrt(512)

    if (tid < 16) { sm[tid] = -3.4e38f; sl[tid] = 0.f; }
    __syncthreads();

    const unsigned short* qrow = Qb + (unsigned)(m0 + nloc) * DD;  // A row for S

    v8f accO[4] = {};   // wave owns output cols [wave*64, wave*64+64)

    for (int kb = 0; kb < NN; kb += 128) {
        // warm L2/WGP$ for the next K chunk (global_prefetch_b8)
        if (kb + 128 < NN)
            __builtin_prefetch(Kb + (unsigned)(kb + 128) * DD + tid * 64, 0, 3);

        // ---- phase 1: S tile (cols wave*16..+15) = Q @ K^T ----
        v8f s = {};
        for (int kd = 0; kd < DD; kd += 32) {
            BF16Frag a, b;
            a.h[0] = *(const v8u*)(qrow + kd + 16 * h);
            a.h[1] = *(const v8u*)(qrow + kd + 16 * h + 8);
            // B column n = key row (kb + wave*16 + nloc): contiguous along d
            b.u = *(const v16u*)(Kb + (unsigned)(kb + wave * 16 + nloc) * DD + kd + 16 * h);
            s = wmma_bf16(a.b, b.b, s);
        }
#pragma unroll
        for (int j = 0; j < 8; ++j)
            sS[(j + 8 * h) * 128 + wave * 16 + nloc] = s[j];
        __syncthreads();

        // ---- phase 2: running row max / exp / row sum, all 256 threads ----
        {
            const int row = tid >> 4;       // 16 threads per row
            const int sub = tid & 15;       // each handles 8 columns
            float* rowS = sS + row * 128 + sub * 8;

            float lm = -3.4e38f;
#pragma unroll
            for (int c = 0; c < 8; ++c) lm = fmaxf(lm, rowS[c]);
            sredm[row][sub] = lm;
            __syncthreads();

            const float mold = sm[row];
            float mnew = mold;
#pragma unroll
            for (int s2 = 0; s2 < 16; ++s2) mnew = fmaxf(mnew, sredm[row][s2]);

            float lsum = 0.f;
#pragma unroll
            for (int c = 0; c < 8; ++c) {
                float p = __expf((rowS[c] - mnew) * rs);
                sP[row * 128 + sub * 8 + c] = f2bf(p);
                lsum += p;
            }
            sreds[row][sub] = lsum;
            __syncthreads();

            if (sub == 0) {
                float sum = 0.f;
#pragma unroll
                for (int s2 = 0; s2 < 16; ++s2) sum += sreds[row][s2];
                float scale = __expf((mold - mnew) * rs);
                sl[row] = sl[row] * scale + sum;
                sm[row] = mnew;
                sscale[row] = scale;
            }
            __syncthreads();
        }

        // ---- phase 3: O = O*scale + P @ V ----
        float scl[8];
#pragma unroll
        for (int j = 0; j < 8; ++j) scl[j] = sscale[j + 8 * h];
#pragma unroll
        for (int t = 0; t < 4; ++t)
#pragma unroll
            for (int j = 0; j < 8; ++j) accO[t][j] *= scl[j];

        for (int kk = 0; kk < 128; kk += 32) {
            BF16Frag a;
            a.h[0] = *(const v8u*)(sP + nloc * 128 + kk + 16 * h);
            a.h[1] = *(const v8u*)(sP + nloc * 128 + kk + 16 * h + 8);
#pragma unroll
            for (int t = 0; t < 4; ++t) {
                BF16Frag b;  // V^T[col][key..] contiguous 32B per lane
                b.u = *(const v16u*)(Vt + (unsigned)(wave * 64 + t * 16 + nloc) * NN
                                        + kb + kk + 16 * h);
                accO[t] = wmma_bf16(a.b, b.b, accO[t]);
            }
        }
        __syncthreads();
    }

    // ---- epilogue: normalize and store fp32 ----
    float lrec[8];
#pragma unroll
    for (int j = 0; j < 8; ++j) lrec[j] = 1.0f / (sl[j + 8 * h] + 1e-8f);
#pragma unroll
    for (int t = 0; t < 4; ++t)
#pragma unroll
        for (int j = 0; j < 8; ++j)
            out[(unsigned)(m0 + j + 8 * h) * DD + wave * 64 + t * 16 + nloc]
                = accO[t][j] * lrec[j];
}

// ---------------------------------------------------------------------------
extern "C" void kernel_launch(void* const* d_in, const int* in_sizes, int n_in,
                              void* d_out, int out_size, void* d_ws, size_t ws_size,
                              hipStream_t stream) {
    const float* x  = (const float*)d_in[0];
    const float* wq = (const float*)d_in[1];
    const float* wk = (const float*)d_in[2];
    const float* wv = (const float*)d_in[3];
    unsigned short* ws = (unsigned short*)d_ws;
    float* out = (float*)d_out;

    cvt_kernel<<<2048, 256, 0, stream>>>(x, wq, wk, wv, ws);

    dim3 g1(NN / 16, DD / 128, 3);
    qkv_kernel<<<g1, 256, 0, stream>>>(ws);

    flash_kernel<<<NN / 16, 256, 0, stream>>>(ws, out);
}
